// GatedMultimodalFusion_84542136255070
// MI455X (gfx1250) — matrix operand order repaired and checked by reading, not compiled, not run on hardware
//
#include <hip/hip_runtime.h>
#include <math.h>
#include <stdint.h>

// ---------- types ----------
typedef __attribute__((ext_vector_type(16))) __bf16     v16bf;
typedef __attribute__((ext_vector_type(2)))  __bf16     v2bf;
typedef __attribute__((ext_vector_type(8)))  float      v8f;
typedef __attribute__((ext_vector_type(4)))  unsigned   u32x4;
typedef __attribute__((ext_vector_type(2)))  unsigned   u32x2;
typedef __attribute__((ext_vector_type(4)))  float      f32x4;
typedef __attribute__((ext_vector_type(8)))  int        i32x8;
typedef __attribute__((ext_vector_type(4)))  int        i32x4;

union FragAB { v16bf v; u32x4 q[2]; };
union FragC  { v8f   v; float f[8]; };

#if __has_builtin(__builtin_amdgcn_tensor_load_to_lds) && __has_builtin(__builtin_amdgcn_s_wait_tensorcnt)
#define USE_TDM 1
#else
#define USE_TDM 0
#endif

// native bf16 conversion: backend emits v_cvt_pk_bf16_f32 (confirmed in round-2 asm)
__device__ __forceinline__ unsigned short f2bf(float f) {
  return __builtin_bit_cast(unsigned short, (__bf16)f);
}
__device__ __forceinline__ unsigned pk2bf(float lo, float hi) {
  v2bf t; t.x = (__bf16)lo; t.y = (__bf16)hi;
  return __builtin_bit_cast(unsigned, t);
}

// ---------- GEMM: C[M,N] = A[M,K](f32) @ WtT[K,N](bf16) + bias, f32 out ----------
// Block tile 256x128, BK=32, 8 waves as 4(M) x 2(N), each wave computes 64x64:
// 16 WMMAs per 8 ds_load_b128 (A frags) + 64 16-bit LDS loads (B frags).
#define BM 256
#define BN 128
#define BK 32
#define LDA 40    // BK + 8 pad (ushort units) -> conflict-free A frag reads
#define LDB 136   // BN + 8 pad; 272B row stride == TDM pad 4 dwords / 64 dwords

#if USE_TDM
// Issue a TDM 2D tile load: 32 rows x 128 bf16 cols from src (row stride N elems)
// into LDS at lds_off with 16B pad after each 256B row. Wave-uniform args.
__device__ __forceinline__ void tdm_load_b(const unsigned short* src, int N, int K,
                                           unsigned lds_off) {
  const unsigned long long ga = (unsigned long long)(uintptr_t)src;
  u32x4 g0;
  g0.x = 1u;                                                  // count=1, user mode
  g0.y = lds_off;                                             // D#.lds_addr (bytes)
  g0.z = (unsigned)ga;                                        // global_addr[31:0]
  g0.w = (unsigned)((ga >> 32) & 0x01FFFFFFu) | (2u << 30);   // addr[56:32] | type=2
  i32x8 g1;
  g1[0] = (1 << 16)      // data_size = 2 bytes
        | (1 << 20)      // pad_enable
        | (5 << 22)      // pad_interval: 64 dwords (256B) between pads
        | (3 << 25);     // pad_amount: 4 dwords (16B)
  g1[1] = (int)(((unsigned)N & 0xFFFFu) << 16);               // tensor_dim0[15:0]
  g1[2] = (int)((((unsigned)N >> 16) & 0xFFFFu) | (((unsigned)K & 0xFFFFu) << 16));
  g1[3] = (int)((((unsigned)K >> 16) & 0xFFFFu) | (128u << 16)); // tile_dim0 = 128
  g1[4] = 32;                                                 // tile_dim1=32, tile_dim2=0
  g1[5] = N;                                                  // tensor_dim0_stride[31:0]
  g1[6] = 0;
  g1[7] = 0;
  i32x4 gz = {0, 0, 0, 0};
#if __clang_major__ >= 23
  i32x8 gz8 = {0, 0, 0, 0, 0, 0, 0, 0};
  __builtin_amdgcn_tensor_load_to_lds(g0, g1, gz, gz, gz8, 0);
#else
  __builtin_amdgcn_tensor_load_to_lds(g0, g1, gz, gz, 0);
#endif
}
#endif

__global__ __launch_bounds__(256) void gemm_bf16_kernel(
    const float* __restrict__ A, const unsigned short* __restrict__ Wt,
    const float* __restrict__ bias, float* __restrict__ C,
    int M, int N, int K, int ldc, int col0)
{
  __shared__ unsigned short sA[2][BM * LDA];   // 2 * 256*40*2B = 40 KB
  __shared__ unsigned short sB[2][BK * LDB];   // 2 *  32*136*2B = 17 KB

  const int tid  = threadIdx.x;
  const int lane = tid & 31;
  const int wave = tid >> 5;
  const int bm0  = blockIdx.y * BM;
  const int bn0  = blockIdx.x * BN;

  const int wm = (wave & 3) * 64;   // 4 waves along M, 64 rows each
  const int wn = (wave >> 2) * 64;  // 2 waves along N, 64 cols each

  const int a_row = tid >> 3;          // A: 8 thr/row * 4 floats, 8 passes of 32 rows
  const int a_col = (tid & 7) * 4;
  const int b_row = tid >> 4;          // B fallback: 16 thr/row * 8 bf16, 2 passes
  const int b_col = (tid & 15) * 8;

  FragC acc[4][4];
#pragma unroll
  for (int i = 0; i < 4; ++i)
#pragma unroll
    for (int j = 0; j < 4; ++j)
#pragma unroll
      for (int r = 0; r < 8; ++r) acc[i][j].f[r] = 0.f;

  const int nk = K / BK;

  // --- split loaders: fetch (global->regs) early, store (regs->LDS) after math ---
  f32x4 areg[8];
  auto fetch_a = [&](int kb) {
    const int k0 = kb * BK;
#pragma unroll
    for (int p = 0; p < 8; ++p) {
      const int r = a_row + p * 32;
      areg[p] = *(const f32x4*)(A + (size_t)(bm0 + r) * K + (k0 + a_col));
    }
  };
  auto store_a = [&](int buf) {
#pragma unroll
    for (int p = 0; p < 8; ++p) {
      const int r = a_row + p * 32;
      u32x2 u;
      u.x = pk2bf(areg[p].x, areg[p].y);
      u.y = pk2bf(areg[p].z, areg[p].w);
      *(u32x2*)(&sA[buf][r * LDA + a_col]) = u;
    }
  };
#if !USE_TDM
  u32x4 breg[2];
  auto fetch_b = [&](int kb) {
    const int k0 = kb * BK;
#pragma unroll
    for (int p = 0; p < 2; ++p) {
      const int r = b_row + p * 16;
      breg[p] = *(const u32x4*)(Wt + (size_t)(k0 + r) * N + (bn0 + b_col));
    }
  };
  auto store_b = [&](int buf) {
#pragma unroll
    for (int p = 0; p < 2; ++p) {
      const int r = b_row + p * 16;
      *(u32x4*)(&sB[buf][r * LDB + b_col]) = breg[p];
    }
  };
#endif

  // ---- prologue: tile 0 ----
  fetch_a(0);
#if USE_TDM
  if (wave == 0)
    tdm_load_b(Wt + bn0, N, K, (unsigned)(uintptr_t)&sB[0][0]);
#else
  fetch_b(0);
#endif
  store_a(0);
#if USE_TDM
  if (wave == 0) __builtin_amdgcn_s_wait_tensorcnt(0);
#else
  store_b(0);
#endif
  __syncthreads();

  for (int kb = 0; kb < nk; ++kb) {
    const int cur = kb & 1;
    const bool has_next = (kb + 1) < nk;

    if (has_next) {
      fetch_a(kb + 1);                 // global loads in flight during WMMA
#if USE_TDM
      if (wave == 0)
        tdm_load_b(Wt + (size_t)(kb + 1) * BK * N + bn0, N, K,
                   cur ? (unsigned)(uintptr_t)&sB[0][0]
                       : (unsigned)(uintptr_t)&sB[1][0]);
#else
      fetch_b(kb + 1);
#endif
    }

    {
      const int l16 = lane & 15;
      const int akb = (lane >> 4) * 8;   // ISA 16-bit A frag: K base 0 / 8
      FragAB afr[4], bfr[4];
#pragma unroll
      for (int i = 0; i < 4; ++i) {
        const unsigned short* pa = &sA[cur][(wm + i * 16 + l16) * LDA + akb];
        afr[i].q[0] = *(const u32x4*)(pa);        // K = akb .. akb+7
        afr[i].q[1] = *(const u32x4*)(pa + 16);   // K = akb+16 .. akb+23
      }
#pragma unroll
      for (int j = 0; j < 4; ++j) {               // B frag: lane = N col, K same order as A
        const unsigned short* pb = &sB[cur][akb * LDB + wn + j * 16 + l16];
#pragma unroll
        for (int s = 0; s < 16; ++s)
          bfr[j].v[s] = __builtin_bit_cast(__bf16, pb[(s + (s >= 8 ? 8 : 0)) * LDB]);
      }
#pragma unroll
      for (int i = 0; i < 4; ++i)
#pragma unroll
        for (int j = 0; j < 4; ++j)
          acc[i][j].v = __builtin_amdgcn_wmma_f32_16x16x32_bf16(
              false, afr[i].v, false, bfr[j].v, (short)0, acc[i][j].v, false, false);
    }

    if (has_next) {
      store_a(cur ^ 1);                // waits for global loads *after* the math
#if !USE_TDM
      store_b(cur ^ 1);
#endif
    }
#if USE_TDM
    if (wave == 0) __builtin_amdgcn_s_wait_tensorcnt(0);
#endif
    __syncthreads();
  }

  // epilogue: C/D layout -> lanes 0-15: M=r, lanes 16-31: M=8+r; N = lane&15
  const int l16 = lane & 15;
  const int mh  = (lane >> 4) * 8;
#pragma unroll
  for (int j = 0; j < 4; ++j) {
    const int n  = bn0 + wn + j * 16 + l16;
    const float bv = bias[n];
#pragma unroll
    for (int i = 0; i < 4; ++i) {
      const int m0 = bm0 + wm + i * 16 + mh;
#pragma unroll
      for (int r = 0; r < 8; ++r)
        C[(size_t)(m0 + r) * ldc + col0 + n] = acc[i][j].f[r] + bv;
    }
  }
}

// ---------- LayerNorm(+GELU) in place, row width 512, one wave per row ----------
__global__ __launch_bounds__(256) void ln_gelu_kernel(
    float* __restrict__ X, const float* __restrict__ g, const float* __restrict__ b, int M)
{
  const int lane = threadIdx.x & 31;
  const int wave = threadIdx.x >> 5;
  const int row  = blockIdx.x * 8 + wave;
  if (row >= M) return;
  float* x = X + (size_t)row * 512;

  float v[16];
  float s = 0.f;
#pragma unroll
  for (int i = 0; i < 4; ++i) {
    const f32x4 f = *(const f32x4*)(x + i * 128 + lane * 4);
    v[i*4+0] = f.x; v[i*4+1] = f.y; v[i*4+2] = f.z; v[i*4+3] = f.w;
    s += f.x + f.y + f.z + f.w;
  }
#pragma unroll
  for (int o = 16; o >= 1; o >>= 1) s += __shfl_xor(s, o, 32);
  const float mu = s * (1.f / 512.f);
  float q = 0.f;
#pragma unroll
  for (int i = 0; i < 16; ++i) { const float d = v[i] - mu; q += d * d; }
#pragma unroll
  for (int o = 16; o >= 1; o >>= 1) q += __shfl_xor(q, o, 32);
  const float inv = rsqrtf(q * (1.f / 512.f) + 1e-5f);
#pragma unroll
  for (int i = 0; i < 4; ++i) {
    const int base = i * 128 + lane * 4;
    f32x4 f;
#pragma unroll
    for (int c = 0; c < 4; ++c) {
      const int idx = base + c;
      const float y = (v[i*4+c] - mu) * inv * g[idx] + b[idx];
      ((float*)&f)[c] = 0.5f * y * (1.f + erff(y * 0.70710678f));   // exact GELU
    }
    *(f32x4*)(x + base) = f;
  }
}

// ---------- final: out = LN(X)*g+b + r1 + r2 ----------
__global__ __launch_bounds__(256) void ln_residual_kernel(
    const float* __restrict__ X, const float* __restrict__ g, const float* __restrict__ b,
    const float* __restrict__ r1, const float* __restrict__ r2,
    float* __restrict__ out, int M)
{
  const int lane = threadIdx.x & 31;
  const int wave = threadIdx.x >> 5;
  const int row  = blockIdx.x * 8 + wave;
  if (row >= M) return;
  const size_t off = (size_t)row * 512;
  const float* x = X + off;

  float v[16];
  float s = 0.f;
#pragma unroll
  for (int i = 0; i < 4; ++i) {
    const f32x4 f = *(const f32x4*)(x + i * 128 + lane * 4);
    v[i*4+0] = f.x; v[i*4+1] = f.y; v[i*4+2] = f.z; v[i*4+3] = f.w;
    s += f.x + f.y + f.z + f.w;
  }
#pragma unroll
  for (int o = 16; o >= 1; o >>= 1) s += __shfl_xor(s, o, 32);
  const float mu = s * (1.f / 512.f);
  float q = 0.f;
#pragma unroll
  for (int i = 0; i < 16; ++i) { const float d = v[i] - mu; q += d * d; }
#pragma unroll
  for (int o = 16; o >= 1; o >>= 1) q += __shfl_xor(q, o, 32);
  const float inv = rsqrtf(q * (1.f / 512.f) + 1e-5f);
#pragma unroll
  for (int i = 0; i < 4; ++i) {
    const int base = i * 128 + lane * 4;
    f32x4 f;
#pragma unroll
    for (int c = 0; c < 4; ++c) {
      const int idx = base + c;
      const float y = (v[i*4+c] - mu) * inv * g[idx] + b[idx];
      ((float*)&f)[c] = y + r1[off + idx] + r2[off + idx];
    }
    *(f32x4*)(out + off + base) = f;
  }
}

// ---------- elementwise gate: P *= sigmoid(G) ----------
__global__ __launch_bounds__(256) void sigmul_kernel(
    float* __restrict__ P, const float* __restrict__ G, int n)
{
  int i = blockIdx.x * blockDim.x + threadIdx.x;
  const int stride = gridDim.x * blockDim.x;
  for (; i < n; i += stride)
    P[i] = P[i] * (1.f / (1.f + __expf(-G[i])));
}

// ---------- weight prep: Wt[k*N+n] = bf16(W[n*K+k]) ----------
__global__ __launch_bounds__(256) void wtrans_kernel(
    const float* __restrict__ W, unsigned short* __restrict__ Wt, int N, int K)
{
  const int idx = blockIdx.x * blockDim.x + threadIdx.x;
  if (idx >= N * K) return;
  const int k = idx / N;
  const int n = idx - k * N;
  Wt[idx] = f2bf(W[(size_t)n * K + k]);
}

// ---------- folded attention weight: WatT[k*512+n] = bf16( (Wout @ Wv)[n,k] ) ----------
__global__ __launch_bounds__(256) void watt_kernel(
    const float* __restrict__ Wout, const float* __restrict__ Win,
    unsigned short* __restrict__ WatT)
{
  const int idx = blockIdx.x * blockDim.x + threadIdx.x;   // 512*512
  const int n = idx & 511, k = idx >> 9;
  float acc = 0.f;
  for (int j = 0; j < 512; ++j)
    acc += Wout[n * 512 + j] * Win[(size_t)(1024 + j) * 512 + k];   // Wv = Win[2D:3D]
  WatT[k * 512 + n] = f2bf(acc);
}

__global__ __launch_bounds__(256) void batt_kernel(
    const float* __restrict__ Wout, const float* __restrict__ binp,
    const float* __restrict__ bout, float* __restrict__ b_att)
{
  const int n = blockIdx.x * blockDim.x + threadIdx.x;
  if (n >= 512) return;
  float acc = bout[n];
  for (int j = 0; j < 512; ++j) acc += Wout[n * 512 + j] * binp[1024 + j];
  b_att[n] = acc;
}

// ---------- host ----------
extern "C" void kernel_launch(void* const* d_in, const int* in_sizes, int n_in,
                              void* d_out, int out_size, void* d_ws, size_t ws_size,
                              hipStream_t stream)
{
  (void)in_sizes; (void)n_in; (void)out_size; (void)ws_size;
  const float* img  = (const float*)d_in[0];
  const float* tab  = (const float*)d_in[1];
  const float* Wi   = (const float*)d_in[2];
  const float* bi   = (const float*)d_in[3];
  const float* lnig = (const float*)d_in[4];
  const float* lnib = (const float*)d_in[5];
  const float* Wtab = (const float*)d_in[6];
  const float* bt   = (const float*)d_in[7];
  const float* lntg = (const float*)d_in[8];
  const float* lntb = (const float*)d_in[9];
  const float* Wgi  = (const float*)d_in[10];
  const float* bgi  = (const float*)d_in[11];
  const float* Wgt  = (const float*)d_in[12];
  const float* bgt  = (const float*)d_in[13];
  const float* Win  = (const float*)d_in[14];
  const float* binp = (const float*)d_in[15];
  const float* Wout = (const float*)d_in[16];
  const float* bout = (const float*)d_in[17];
  const float* Wf1  = (const float*)d_in[18];
  const float* bf1  = (const float*)d_in[19];
  const float* lf1g = (const float*)d_in[20];
  const float* lf1b = (const float*)d_in[21];
  const float* Wf2  = (const float*)d_in[22];
  const float* bf2  = (const float*)d_in[23];
  const float* lf2g = (const float*)d_in[24];
  const float* lf2b = (const float*)d_in[25];
  float* out = (float*)d_out;

  const int Bn = 16384;

  char* p = (char*)d_ws;
  auto alloc = [&](size_t bytes) -> void* {
    void* r = (void*)p;
    p += (bytes + 255) & ~(size_t)255;
    return r;
  };
  unsigned short* WiT  = (unsigned short*)alloc((size_t)2048 * 512 * 2);
  unsigned short* WtT  = (unsigned short*)alloc((size_t)128  * 512 * 2);
  unsigned short* WgiT = (unsigned short*)alloc((size_t)512  * 512 * 2);
  unsigned short* WgtT = (unsigned short*)alloc((size_t)512  * 512 * 2);
  unsigned short* WatT = (unsigned short*)alloc((size_t)512  * 512 * 2);
  unsigned short* Wf1T = (unsigned short*)alloc((size_t)1024 * 512 * 2);
  unsigned short* Wf2T = (unsigned short*)alloc((size_t)512  * 512 * 2);
  float* b_att = (float*)alloc(512 * 4);
  float* bufPi = (float*)alloc((size_t)Bn * 512 * 4);   // img_proj -> img_gated
  float* bufPt = (float*)alloc((size_t)Bn * 512 * 4);   // tab_proj -> tab_gated
  float* bufG  = (float*)alloc((size_t)Bn * 512 * 4);   // lin scratch / h
  float* bufC  = (float*)alloc((size_t)Bn * 1024 * 4);  // combined att; reused for f2_lin

  const dim3 blk(256);
  // weight prep (tiny, one-time per launch)
  wtrans_kernel<<<(2048*512 + 255) / 256, blk, 0, stream>>>(Wi,  WiT,  512, 2048);
  wtrans_kernel<<<(128*512  + 255) / 256, blk, 0, stream>>>(Wtab, WtT, 512, 128);
  wtrans_kernel<<<(512*512  + 255) / 256, blk, 0, stream>>>(Wgi, WgiT, 512, 512);
  wtrans_kernel<<<(512*512  + 255) / 256, blk, 0, stream>>>(Wgt, WgtT, 512, 512);
  wtrans_kernel<<<(512*1024 + 255) / 256, blk, 0, stream>>>(Wf1, Wf1T, 512, 1024);
  wtrans_kernel<<<(512*512  + 255) / 256, blk, 0, stream>>>(Wf2, Wf2T, 512, 512);
  watt_kernel<<<(512*512) / 256, blk, 0, stream>>>(Wout, Win, WatT);
  batt_kernel<<<2, blk, 0, stream>>>(Wout, binp, bout, b_att);

  const dim3 g512(512 / BN, Bn / BM);   // (4, 64)

  // proj branches
  gemm_bf16_kernel<<<g512, blk, 0, stream>>>(img, WiT, bi, bufPi, Bn, 512, 2048, 512, 0);
  ln_gelu_kernel<<<Bn / 8, blk, 0, stream>>>(bufPi, lnig, lnib, Bn);
  gemm_bf16_kernel<<<g512, blk, 0, stream>>>(tab, WtT, bt, bufPt, Bn, 512, 128, 512, 0);
  ln_gelu_kernel<<<Bn / 8, blk, 0, stream>>>(bufPt, lntg, lntb, Bn);
  // gates (in-place: proj buffers become gated buffers)
  gemm_bf16_kernel<<<g512, blk, 0, stream>>>(bufPi, WgiT, bgi, bufG, Bn, 512, 512, 512, 0);
  sigmul_kernel<<<4096, blk, 0, stream>>>(bufPi, bufG, Bn * 512);
  gemm_bf16_kernel<<<g512, blk, 0, stream>>>(bufPt, WgtT, bgt, bufG, Bn, 512, 512, 512, 0);
  sigmul_kernel<<<4096, blk, 0, stream>>>(bufPt, bufG, Bn * 512);
  // len-1 attention, folded: img_att = tab_gated @ Watt^T -> cols [0,512);
  //                          tab_att = img_gated @ Watt^T -> cols [512,1024)
  gemm_bf16_kernel<<<g512, blk, 0, stream>>>(bufPt, WatT, b_att, bufC, Bn, 512, 512, 1024, 0);
  gemm_bf16_kernel<<<g512, blk, 0, stream>>>(bufPi, WatT, b_att, bufC, Bn, 512, 512, 1024, 512);
  // fusion MLP
  gemm_bf16_kernel<<<g512, blk, 0, stream>>>(bufC, Wf1T, bf1, bufG, Bn, 512, 1024, 512, 0);
  ln_gelu_kernel<<<Bn / 8, blk, 0, stream>>>(bufG, lf1g, lf1b, Bn);
  gemm_bf16_kernel<<<g512, blk, 0, stream>>>(bufG, Wf2T, bf2, bufC, Bn, 512, 512, 512, 0);
  ln_residual_kernel<<<Bn / 8, blk, 0, stream>>>(bufC, lf2g, lf2b, bufPi, bufPt, out, Bn);
}
